// LSTMClassifier_51161650430049
// MI455X (gfx1250) — matrix operand order repaired and checked by reading, not compiled, not run on hardware
//
#include <hip/hip_runtime.h>

#define TLEN 500
#define BATCH 2048
#define H1C 128
#define H2C 64

typedef __attribute__((ext_vector_type(16))) _Float16 v16h;
typedef __attribute__((ext_vector_type(8)))  _Float16 v8h;
typedef __attribute__((ext_vector_type(8)))  float    v8f;

__device__ __forceinline__ float sigmf(float x) { return 1.0f / (1.0f + __expf(-x)); }
__device__ __forceinline__ float tanhfast(float x) {
  float e = __expf(-2.0f * x);
  return (1.0f - e) / (1.0f + e);
}

__device__ __forceinline__ v16h frag_cat(v8h a, v8h b) {
  return __builtin_shufflevector(a, b, 0,1,2,3,4,5,6,7,8,9,10,11,12,13,14,15);
}

// A fragment: H is (16 x ldk) row-major fp16; 16x32 tile starting at k0.
// lanes 0-15 -> row=lane, K chunks [k0,k0+8)+[k0+16,k0+24);
// lanes 16-31 -> row=lane-16, K chunks [k0+8,k0+16)+[k0+24,k0+32).
__device__ __forceinline__ v16h load_a(const _Float16* H, int ldk, int k0, int lane) {
  const int m = lane & 15;
  const int sel8 = (lane & 16) ? 8 : 0;
  const _Float16* p = H + m * ldk + k0 + sel8;
  v8h a = *(const v8h*)(p);
  v8h b = *(const v8h*)(p + 16);
  return frag_cat(a, b);
}

// B fragment: W is (N x K) row-major fp16; tile = W^T[k0:k0+32, n0:n0+16].
// lanes 0-15 -> col n, K chunks [k0,k0+8)+[k0+16,k0+24); lanes 16-31 -> col n, K chunks [k0+8,k0+16)+[k0+24,k0+32).
__device__ __forceinline__ v16h load_b(const _Float16* W, int ldk, int n0, int k0, int lane) {
  const int n = lane & 15;
  const int sel16 = (lane & 16) ? 8 : 0;
  const _Float16* p = W + (size_t)(n0 + n) * ldk + k0 + sel16;
  v8h a = *(const v8h*)(p);
  v8h b = *(const v8h*)(p + 16);
  return frag_cat(a, b);
}

#define WMMA(Af, Bf, Cf) \
  __builtin_amdgcn_wmma_f32_16x16x32_f16(false, (Af), false, (Bf), (short)0, (Cf), false, false)

// ---------------- prep: fold input projection, convert weights to fp16 ----------------
__global__ void prep_kernel(const float* __restrict__ Wp, const float* __restrict__ bp,
                            const float* __restrict__ Wih1, const float* __restrict__ Whh1,
                            const float* __restrict__ bih1, const float* __restrict__ bhh1,
                            const float* __restrict__ Wih2, const float* __restrict__ Whh2,
                            const float* __restrict__ bih2, const float* __restrict__ bhh2,
                            float* __restrict__ vg, float* __restrict__ ug,
                            float* __restrict__ b2g,
                            _Float16* __restrict__ whh1h, _Float16* __restrict__ wih2h,
                            _Float16* __restrict__ whh2h) {
  int i = blockIdx.x * blockDim.x + threadIdx.x;
  if (i < 4 * H1C * H1C) whh1h[i] = (_Float16)Whh1[i];   // 512x128
  if (i < 4 * H2C * H1C) wih2h[i] = (_Float16)Wih2[i];   // 256x128
  if (i < 4 * H2C * H2C) whh2h[i] = (_Float16)Whh2[i];   // 256x64
  if (i < 4 * H1C) {
    float sv = 0.f, su = 0.f;
    for (int j = 0; j < H1C; ++j) {
      float w = Wih1[i * H1C + j];
      sv += w * Wp[j];
      su += w * bp[j];
    }
    vg[i] = sv;
    ug[i] = su + bih1[i] + bhh1[i];
  }
  if (i < 4 * H2C) b2g[i] = bih2[i] + bhh2[i];
}

// ---------------- fused 2-layer LSTM + FC head, persistent per 16-row batch tile ----------------
__global__ void __launch_bounds__(384, 1)
lstm_fused(const float* __restrict__ x,
           const float* __restrict__ vg, const float* __restrict__ ug,
           const float* __restrict__ b2g,
           const _Float16* __restrict__ whh1h,
           const _Float16* __restrict__ wih2h,
           const _Float16* __restrict__ whh2h,
           const float* __restrict__ W1, const float* __restrict__ b1,
           const float* __restrict__ W2, const float* __restrict__ b2,
           float* __restrict__ out) {
  __shared__ _Float16 h1buf[2 * 16 * H1C];  // 8 KB double buffer
  __shared__ _Float16 h2buf[2 * 16 * H2C];  // 4 KB double buffer
  __shared__ float    sXH[16 * TLEN];       // 31.25 KB x slab; reused as h3 after loop

  const int tid  = threadIdx.x;
  const int lane = tid & 31;
  const int wv   = tid >> 5;          // 12 waves: 0-7 LSTM1, 8-11 LSTM2
  const int b0   = blockIdx.x * 16;

  // ---- stage x slab, zero h buffers ----
  for (int idx = tid; idx < 3072; idx += 384)   // (8192+4096)/4 words, buffers contiguous
    ((unsigned int*)h1buf)[idx] = 0u;
  for (int idx = tid; idx < 16 * TLEN; idx += 384) {
    int m = idx / TLEN, t = idx - m * TLEN;     // coalesced read of x rows
    sXH[t * 16 + m] = x[(size_t)(b0 + m) * TLEN + t];
  }

  const int mbase = (lane & 16) ? 8 : 0;  // C/D frag: vgpr r -> row mbase+r, col = lane&15
  const int col   = lane & 15;

  __syncthreads();

  if (wv < 8) {
    // ================= LSTM1 waves: hidden tile wv =================
    v16h B1[4][4];  // [k][q] Whh1 fragments, register-resident across all 500 steps
#pragma unroll
    for (int k = 0; k < 4; ++k)
#pragma unroll
      for (int q = 0; q < 4; ++q)
        B1[k][q] = load_b(whh1h, H1C, q * H1C + wv * 16, k * 32, lane);
    float vq[4], uq[4];  // folded input-projection coefficients (loop-invariant)
#pragma unroll
    for (int q = 0; q < 4; ++q) {
      vq[q] = vg[q * H1C + wv * 16 + col];
      uq[q] = ug[q * H1C + wv * 16 + col];
    }
    float c1[8];
#pragma unroll
    for (int r = 0; r < 8; ++r) c1[r] = 0.f;

    for (int i = 0; i <= TLEN; ++i) {
      if (i < TLEN) {
        const int p = i & 1;
        const _Float16* h1r = h1buf + p * (16 * H1C);        // h1_{i-1}
        _Float16*       h1w = h1buf + (1 - p) * (16 * H1C);  // h1_i
        float xr[8];
#pragma unroll
        for (int r = 0; r < 8; ++r) xr[r] = sXH[i * 16 + mbase + r];
        v8f acc[4] = {};  // inline-zero SRC2 on first WMMA of each chain
        // rotating A double-buffer: issue load for group k+1 before group k's WMMAs
        v16h Ac = load_a(h1r, H1C, 0, lane);
#pragma unroll
        for (int k = 0; k < 4; ++k) {
          v16h An;
          if (k < 3) An = load_a(h1r, H1C, (k + 1) * 32, lane);
#pragma unroll
          for (int q = 0; q < 4; ++q) acc[q] = WMMA(Ac, B1[k][q], acc[q]);
          if (k < 3) Ac = An;
        }
#pragma unroll
        for (int r = 0; r < 8; ++r) {
          float ig = sigmf(fmaf(xr[r], vq[0], uq[0]) + acc[0][r]);
          float fg = sigmf(fmaf(xr[r], vq[1], uq[1]) + acc[1][r]);
          float gg = tanhfast(fmaf(xr[r], vq[2], uq[2]) + acc[2][r]);
          float og = sigmf(fmaf(xr[r], vq[3], uq[3]) + acc[3][r]);
          c1[r] = fg * c1[r] + ig * gg;
          float h = og * tanhfast(c1[r]);
          h1w[(mbase + r) * H1C + wv * 16 + col] = (_Float16)h;
        }
      }
      __syncthreads();
    }
  } else {
    // ================= LSTM2 waves: hidden tile wv-8, one step behind =================
    const int w2 = wv - 8;
    // unified B fragments [k][q]: k=0..3 -> Wih2 (K=128), k=4..5 -> Whh2 (K=64)
    v16h Ball[6][4];
#pragma unroll
    for (int k = 0; k < 4; ++k)
#pragma unroll
      for (int q = 0; q < 4; ++q)
        Ball[k][q] = load_b(wih2h, H1C, q * H2C + w2 * 16, k * 32, lane);
#pragma unroll
    for (int k = 0; k < 2; ++k)
#pragma unroll
      for (int q = 0; q < 4; ++q)
        Ball[4 + k][q] = load_b(whh2h, H2C, q * H2C + w2 * 16, k * 32, lane);
    float bb[4];
#pragma unroll
    for (int q = 0; q < 4; ++q) bb[q] = b2g[q * H2C + w2 * 16 + col];
    float c2[8];
#pragma unroll
    for (int r = 0; r < 8; ++r) c2[r] = 0.f;

    for (int i = 0; i <= TLEN; ++i) {
      if (i >= 1) {
        const int p = i & 1;
        const _Float16* h1r = h1buf + p * (16 * H1C);        // h1_{i-1}
        const _Float16* h2r = h2buf + p * (16 * H2C);        // h2_{i-2}
        _Float16*       h2w = h2buf + (1 - p) * (16 * H2C);  // h2_{i-1}
        v8f acc[4] = {};
        // rotating A double-buffer across all 6 k-groups (4x h1, 2x h2)
        v16h Ac = load_a(h1r, H1C, 0, lane);
#pragma unroll
        for (int k = 0; k < 6; ++k) {
          v16h An;
          if (k < 3)       An = load_a(h1r, H1C, (k + 1) * 32, lane);
          else if (k == 3) An = load_a(h2r, H2C, 0, lane);
          else if (k == 4) An = load_a(h2r, H2C, 32, lane);
#pragma unroll
          for (int q = 0; q < 4; ++q) acc[q] = WMMA(Ac, Ball[k][q], acc[q]);
          if (k < 5) Ac = An;
        }
#pragma unroll
        for (int r = 0; r < 8; ++r) {
          float ig = sigmf(acc[0][r] + bb[0]);
          float fg = sigmf(acc[1][r] + bb[1]);
          float gg = tanhfast(acc[2][r] + bb[2]);
          float og = sigmf(acc[3][r] + bb[3]);
          c2[r] = fg * c2[r] + ig * gg;
          float h = og * tanhfast(c2[r]);
          h2w[(mbase + r) * H2C + w2 * 16 + col] = (_Float16)h;
        }
      }
      __syncthreads();
    }
  }

  // ---- FC head (tiny): h2_final lives in h2buf[1 - (TLEN&1)] ----
  const _Float16* h2f = h2buf + (1 - (TLEN & 1)) * (16 * H2C);
  float* sH3 = (float*)sXH;  // reuse x slab (needs 8 KB of 31 KB)
  for (int idx = tid; idx < 16 * 128; idx += 384) {
    int m = idx >> 7, n = idx & 127;
    float s = b1[n];
#pragma unroll 4
    for (int k = 0; k < H2C; ++k) s += (float)h2f[m * H2C + k] * W1[n * H2C + k];
    sH3[idx] = fmaxf(s, 0.f);
  }
  __syncthreads();
  for (int idx = tid; idx < 16 * 4; idx += 384) {
    int m = idx >> 2, n = idx & 3;
    float s = b2[n];
#pragma unroll 4
    for (int k = 0; k < 128; ++k) s += sH3[m * 128 + k] * W2[n * 128 + k];
    out[(size_t)(b0 + m) * 4 + n] = s;
  }
}

extern "C" void kernel_launch(void* const* d_in, const int* in_sizes, int n_in,
                              void* d_out, int out_size, void* d_ws, size_t ws_size,
                              hipStream_t stream) {
  const float* x    = (const float*)d_in[0];
  const float* Wp   = (const float*)d_in[1];
  const float* bp   = (const float*)d_in[2];
  const float* Wih1 = (const float*)d_in[3];
  const float* Whh1 = (const float*)d_in[4];
  const float* bih1 = (const float*)d_in[5];
  const float* bhh1 = (const float*)d_in[6];
  const float* Wih2 = (const float*)d_in[7];
  const float* Whh2 = (const float*)d_in[8];
  const float* bih2 = (const float*)d_in[9];
  const float* bhh2 = (const float*)d_in[10];
  const float* W1   = (const float*)d_in[11];
  const float* b1   = (const float*)d_in[12];
  const float* W2   = (const float*)d_in[13];
  const float* b2   = (const float*)d_in[14];
  float* out = (float*)d_out;

  char* ws = (char*)d_ws;
  float*    vg    = (float*)(ws + 0);         //  2048 B
  float*    ug    = (float*)(ws + 2048);      //  2048 B
  float*    b2g   = (float*)(ws + 4096);      //  1024 B (pad to 8192)
  _Float16* whh1h = (_Float16*)(ws + 8192);   // 131072 B
  _Float16* wih2h = (_Float16*)(ws + 139264); //  65536 B
  _Float16* whh2h = (_Float16*)(ws + 204800); //  32768 B

  prep_kernel<<<256, 256, 0, stream>>>(Wp, bp, Wih1, Whh1, bih1, bhh1,
                                       Wih2, Whh2, bih2, bhh2,
                                       vg, ug, b2g, whh1h, wih2h, whh2h);

  lstm_fused<<<BATCH / 16, 384, 0, stream>>>(
      x, vg, ug, b2g, whh1h, wih2h, whh2h, W1, b1, W2, b2, out);
}
